// M3Care_78400333021298
// MI455X (gfx1250) — matrix operand order repaired and checked by reading, not compiled, not run on hardware
//
#include <hip/hip_runtime.h>

#define DEV __device__ __forceinline__

typedef __attribute__((ext_vector_type(16))) __bf16 bf16x16;
typedef __attribute__((ext_vector_type(8)))  __bf16 bf16x8;
typedef __attribute__((ext_vector_type(8)))  float  f32x8;

static constexpr int  BATCH = 4096;
static constexpr int  HDIM  = 128;
static constexpr long BH    = (long)BATCH * HDIM;
static constexpr int  TOK   = BATCH * 5;
static constexpr long TOKH  = (long)TOK * HDIM;

// ---------------------------------------------------------------- helpers
DEV float block_reduce(float v, float* red, int tid, int n) {
  red[tid] = v; __syncthreads();
  for (int s = n >> 1; s > 0; s >>= 1) {
    if (tid < s) red[tid] += red[tid + s];
    __syncthreads();
  }
  float r = red[0];
  __syncthreads();
  return r;
}

DEV __bf16 tobf(float x)  { return (__bf16)x; }
DEV __bf16 tobf(__bf16 x) { return x; }

// A-matrix 16x32 bf16 fragment from LDS (row-major [m][k], stride in elems).
// lane: m = lane&15, hi = lane>>4; elems 0..7 -> k = 8*hi + e, elems 8..15 -> k = 16 + 8*hi + (e-8)
DEV bf16x16 frag_a(const __bf16* base, int stride) {
  int lane = threadIdx.x & 31;
  int m = lane & 15, hi = lane >> 4;
  const __bf16* p = base + m * stride + hi * 8;
  bf16x8 lo = *(const bf16x8*)p;
  bf16x8 hv = *(const bf16x8*)(p + 16);
  bf16x16 r;
#pragma unroll
  for (int i = 0; i < 8; ++i) { r[i] = lo[i]; r[i + 8] = hv[i]; }
  return r;
}

// B-matrix 32x16 bf16 fragment from LDS stored transposed ([n][k]).
// lane: n = lane&15, hi = lane>>4; elems 0..7 -> k = 8*hi + e, elems 8..15 -> k = 16 + 8*hi + (e-8)
DEV bf16x16 frag_b(const __bf16* base, int stride) {
  int lane = threadIdx.x & 31;
  int n = lane & 15, hi = lane >> 4;
  const __bf16* p = base + n * stride + hi * 8;
  bf16x8 lo = *(const bf16x8*)p;
  bf16x8 hv = *(const bf16x8*)(p + 16);
  bf16x16 r;
#pragma unroll
  for (int i = 0; i < 8; ++i) { r[i] = lo[i]; r[i + 8] = hv[i]; }
  return r;
}

DEV f32x8 wmma_bf16(bf16x16 a, bf16x16 b, f32x8 c) {
  return __builtin_amdgcn_wmma_f32_16x16x32_bf16(false, a, false, b, (short)0, c, false, false);
}

// ---------------------------------------------------------------- generic WMMA GEMM
// C[M,N] = act(A[M,K] @ W[K,N] + bias), 64x64 tile, 128 threads (4 waves)
template<typename AT, typename WT, int ACT>
__global__ __launch_bounds__(128)
void gemm_wmma_kernel(const AT* __restrict__ A, const WT* __restrict__ W,
                      const float* __restrict__ bias, float* __restrict__ C,
                      int M, int N, int K) {
  __shared__ __align__(16) __bf16 sA[64][40];
  __shared__ __align__(16) __bf16 sW[64][40]; // transposed: [n][k]
  int m0 = blockIdx.x * 64, n0 = blockIdx.y * 64;
  int tid = threadIdx.x, lane = tid & 31, wid = tid >> 5;
  f32x8 acc[4] = {};
  for (int kc = 0; kc < K; kc += 32) {
#pragma unroll
    for (int i = tid; i < 64 * 32; i += 128) {
      int r = i >> 5, c = i & 31;
      sA[r][c] = tobf(A[(long)(m0 + r) * K + kc + c]);
    }
#pragma unroll
    for (int i = tid; i < 32 * 64; i += 128) {
      int k = i >> 6, n = i & 63;
      sW[n][k] = tobf(W[(long)(kc + k) * N + n0 + n]);
    }
    __syncthreads();
    bf16x16 af = frag_a(&sA[wid * 16][0], 40);
#pragma unroll
    for (int nt = 0; nt < 4; ++nt) {
      bf16x16 bfr = frag_b(&sW[nt * 16][0], 40);
      acc[nt] = wmma_bf16(af, bfr, acc[nt]);
    }
    __syncthreads();
  }
  int nl = lane & 15, hi = lane >> 4;
#pragma unroll
  for (int nt = 0; nt < 4; ++nt) {
    int col = n0 + nt * 16 + nl;
    float bv = bias ? bias[col] : 0.f;
#pragma unroll
    for (int c = 0; c < 8; ++c) {
      int row = m0 + wid * 16 + c + 8 * hi;
      float v = acc[nt][c] + bv;
      if (ACT == 1) v = fmaxf(v, 0.f);
      C[(long)row * N + col] = v;
    }
  }
}

template<typename AT, typename WT, int ACT>
static inline void launch_gemm(const AT* A, const WT* W, const float* bias, float* C,
                               int M, int N, int K, hipStream_t s) {
  dim3 grid(M / 64, N / 64);
  gemm_wmma_kernel<AT, WT, ACT><<<grid, 128, 0, s>>>(A, W, bias, C, M, N, K);
}

// ---------------------------------------------------------------- fused dual-Gram + gaussian sim accumulate
// 128x128 tile, 256 threads (8 waves). Computes G1=Y1 Y1^T, G2=Y2 Y2^T tiles, then
// agg += ((1-e)*K1 + e)*K2 * (mask_i*mask_j)
__global__ __launch_bounds__(256)
void sim_accum_kernel(const __bf16* __restrict__ Y1, const __bf16* __restrict__ Y2,
                      const float* __restrict__ sq1, const float* __restrict__ sq2,
                      const float* __restrict__ bw1, const float* __restrict__ bw2,
                      const float* __restrict__ msk, float* __restrict__ agg) {
  __shared__ __align__(16) __bf16 s1i[128][40];
  __shared__ __align__(16) __bf16 s1j[128][40];
  __shared__ __align__(16) __bf16 s2i[128][40];
  __shared__ __align__(16) __bf16 s2j[128][40];
  int i0 = blockIdx.x * 128, j0 = blockIdx.y * 128;
  int tid = threadIdx.x, lane = tid & 31, wid = tid >> 5;
  f32x8 a1[8] = {}, a2[8] = {};
  for (int kc = 0; kc < HDIM; kc += 32) {
#pragma unroll
    for (int i = tid; i < 128 * 32; i += 256) {
      int r = i >> 5, c = i & 31;
      s1i[r][c] = Y1[(long)(i0 + r) * HDIM + kc + c];
      s1j[r][c] = Y1[(long)(j0 + r) * HDIM + kc + c];
      s2i[r][c] = Y2[(long)(i0 + r) * HDIM + kc + c];
      s2j[r][c] = Y2[(long)(j0 + r) * HDIM + kc + c];
    }
    __syncthreads();
    bf16x16 af1 = frag_a(&s1i[wid * 16][0], 40);
    bf16x16 af2 = frag_a(&s2i[wid * 16][0], 40);
#pragma unroll
    for (int nt = 0; nt < 8; ++nt) {
      bf16x16 b1 = frag_b(&s1j[nt * 16][0], 40);
      a1[nt] = wmma_bf16(af1, b1, a1[nt]);
      bf16x16 b2 = frag_b(&s2j[nt * 16][0], 40);
      a2[nt] = wmma_bf16(af2, b2, a2[nt]);
    }
    __syncthreads();
  }
  float ib10 = bw1[0], ib11 = bw1[1], ib12 = bw1[2], e = bw1[3];
  float ib20 = bw2[0], ib21 = bw2[1], ib22 = bw2[2];
  int nl = lane & 15, hi = lane >> 4;
#pragma unroll
  for (int nt = 0; nt < 8; ++nt) {
    int j = j0 + nt * 16 + nl;
    float q1j = sq1[j], q2j = sq2[j];
    float mj = msk ? msk[j] : 1.f;
#pragma unroll
    for (int c = 0; c < 8; ++c) {
      int i = i0 + wid * 16 + c + 8 * hi;
      float d1 = sq1[i] + q1j - 2.f * a1[nt][c];
      float d2 = sq2[i] + q2j - 2.f * a2[nt][c];
      float K1 = __expf(-d1 * ib10) + __expf(-d1 * ib11) + __expf(-d1 * ib12);
      float K2 = __expf(-d2 * ib20) + __expf(-d2 * ib21) + __expf(-d2 * ib22);
      float sim = ((1.f - e) * K1 + e) * K2;
      if (msk) sim *= msk[i] * mj;
      agg[(long)i * BATCH + j] += sim;
    }
  }
}

// ---------------------------------------------------------------- BN / bandwidth
template<int RELU>
__global__ void bn_stats_kernel(const float* __restrict__ X, float* mean, float* rstd) {
  __shared__ float r1[256], r2[256];
  int c = blockIdx.x, t = threadIdx.x;
  float s = 0.f, s2 = 0.f;
  for (int r = t; r < BATCH; r += 256) {
    float v = X[(long)r * HDIM + c];
    if (RELU) v = fmaxf(v, 0.f);
    s += v; s2 += v * v;
  }
  r1[t] = s; r2[t] = s2; __syncthreads();
  for (int k = 128; k > 0; k >>= 1) {
    if (t < k) { r1[t] += r1[t + k]; r2[t] += r2[t + k]; }
    __syncthreads();
  }
  if (t == 0) {
    float m = r1[0] / (float)BATCH;
    float var = r2[0] / (float)BATCH - m * m;
    mean[c] = m;
    rstd[c] = rsqrtf(var + 1e-5f);
  }
}

template<int RELU>
__global__ void bn_norm_kernel(const float* __restrict__ X, const float* mean, const float* rstd,
                               const float* g, const float* b, __bf16* __restrict__ Y,
                               float* rowsq, float* colsum) {
  __shared__ float red[128];
  int r = blockIdx.x, c = threadIdx.x;
  float v = X[(long)r * HDIM + c];
  if (RELU) v = fmaxf(v, 0.f);
  float y = (v - mean[c]) * rstd[c] * g[c] + b[c];
  Y[(long)r * HDIM + c] = (__bf16)y;
  atomicAdd(&colsum[c], y);
  float tot = block_reduce(y * y, red, c, 128);
  if (c == 0) rowsq[r] = tot;
}

// sum(d) = 2n*sum(rowsq) - 2*||colsum||^2 ; bw = sum(d)/(n^2-n) / GK_MUL^(GK_NUM/2)
__global__ void bw_finalize_kernel(const float* __restrict__ rowsq, const float* __restrict__ colsum,
                                   const float* __restrict__ simi_eps, int eps_idx,
                                   float* __restrict__ out) {
  __shared__ float red[256];
  int t = threadIdx.x;
  float s = 0.f;
  for (int i = t; i < BATCH; i += 256) s += rowsq[i];
  float stot = block_reduce(s, red, t, 256);
  float c = 0.f;
  for (int i = t; i < HDIM; i += 256) c += colsum[i] * colsum[i];
  float ctot = block_reduce(c, red, t, 256);
  if (t == 0) {
    float n = (float)BATCH;
    float sumd = 2.f * n * stot - 2.f * ctot;
    float bw = sumd / (n * n - n);
    bw *= 0.5f; // / GK_MUL^(GK_NUM//2) = /2
    out[0] = 1.f / bw;
    out[1] = 1.f / (2.f * bw);
    out[2] = 1.f / (4.f * bw);
    out[3] = 1.f / (1.f + __expf(-simi_eps[eps_idx]));
  }
}

// ---------------------------------------------------------------- small elementwise kernels
__global__ void zero_kernel(float* p, long n) {
  long i = (long)blockIdx.x * 256 + threadIdx.x;
  if (i < n) p[i] = 0.f;
}

__global__ void cvt_bf16_kernel(const float* __restrict__ x, __bf16* __restrict__ y, long n) {
  long i = (long)blockIdx.x * 256 + threadIdx.x;
  if (i < n) y[i] = (__bf16)x[i];
}

__global__ void mask_kernel(const int* __restrict__ missing, float* nstm, float* ntsm) {
  int i = blockIdx.x * 256 + threadIdx.x;
  if (i < BATCH) {
    nstm[i] = (missing[2 * i + 0] == 0) ? 1.f : 0.f;
    ntsm[i] = (missing[2 * i + 1] == 0) ? 1.f : 0.f;
  }
}

__global__ void mask_emb_kernel(const float* __restrict__ src, const float* __restrict__ m,
                                float* __restrict__ dst) {
  long i = (long)blockIdx.x * 256 + threadIdx.x;
  if (i < BH) dst[i] = src[i] * m[i >> 7];
}

__global__ void finalize_agg_kernel(float* __restrict__ num, __bf16* __restrict__ abf,
                                    const float* __restrict__ nstm, const float* __restrict__ ntsm,
                                    const float* __restrict__ dissim) {
  long idx = (long)blockIdx.x * 256 + threadIdx.x;
  if (idx >= (long)BATCH * BATCH) return;
  int i = (int)(idx >> 12), j = (int)(idx & 4095);
  float den = 3.f + nstm[i] * nstm[j] + ntsm[i] * ntsm[j];
  float a = num[idx] / den;
  float thr = 1.f / (1.f + __expf(-dissim[0]));
  a = (a > thr) ? a : 0.f;
  num[idx] = a;
  abf[idx] = (__bf16)a;
}

__global__ void sumsq_kernel(const float* __restrict__ X, long n, float* acc) {
  __shared__ float red[256];
  float s = 0.f;
  for (long i = (long)blockIdx.x * 256 + threadIdx.x; i < n; i += (long)gridDim.x * 256) {
    float v = X[i]; s += v * v;
  }
  float tot = block_reduce(s, red, threadIdx.x, 256);
  if (threadIdx.x == 0) atomicAdd(acc, tot);
}

__global__ void impute_kernel(const float* __restrict__ emb, const float* __restrict__ aux,
                              const float* __restrict__ msk, const float* __restrict__ aw,
                              const float* __restrict__ ab, float* __restrict__ out) {
  __shared__ float red[128];
  int r = blockIdx.x, c = threadIdx.x;
  float e_ = emb[(long)r * HDIM + c], a_ = aux[(long)r * HDIM + c];
  float se = block_reduce(e_ * aw[c], red, c, 128);
  float so = block_reduce(a_ * aw[c], red, c, 128);
  float s = 1.f / (1.f + __expf(-(se + ab[0])));
  float o = 1.f / (1.f + __expf(-(so + ab[0])));
  s = s / (s + o);
  float m = msk[r];
  out[(long)r * HDIM + c] = m * (s * e_ + (1.f - s) * a_) + (1.f - m) * a_;
}

__global__ void build_z0_kernel(const float* __restrict__ dem, const float* __restrict__ vit,
                                const float* __restrict__ itv, const float* __restrict__ nsti,
                                const float* __restrict__ ntsi, const float* __restrict__ modal,
                                float* __restrict__ z0) {
  long idx = (long)blockIdx.x * 256 + threadIdx.x;
  if (idx >= TOKH) return;
  int h = (int)(idx & 127);
  int s = (int)((idx >> 7) % 5);
  long b = idx / (5 * HDIM);
  const float* src = (s == 0) ? dem : (s == 1) ? vit : (s == 2) ? itv : (s == 3) ? nsti : ntsi;
  z0[idx] = src[b * HDIM + h] + modal[s * HDIM + h];
}

// seq-len-5 attention, one block per sample
template<int HEADS>
__global__ __launch_bounds__(128)
void attn_kernel(const float* __restrict__ Q, const float* __restrict__ K,
                 const float* __restrict__ V, const float* __restrict__ nstm,
                 const float* __restrict__ ntsm, float* __restrict__ O) {
  constexpr int DH = HDIM / HEADS;
  __shared__ float q[5][128], k[5][128], v[5][128];
  __shared__ float sc[HEADS][5][5];
  __shared__ float mask[5];
  int b = blockIdx.x, t = threadIdx.x;
  for (int i = t; i < 5 * 128; i += 128) {
    int s = i >> 7, h = i & 127;
    long off = ((long)b * 5 + s) * HDIM + h;
    q[s][h] = Q[off]; k[s][h] = K[off]; v[s][h] = V[off];
  }
  if (t < 5) mask[t] = (t < 3) ? 1.f : ((t == 3) ? nstm[b] : ntsm[b]);
  __syncthreads();
  float scale = rsqrtf((float)DH);
  if (t < HEADS * 25) {
    int hd = t / 25, rem = t % 25, qi = rem / 5, ki = rem % 5;
    float d = 0.f;
    for (int x = 0; x < DH; ++x) d += q[qi][hd * DH + x] * k[ki][hd * DH + x];
    d *= scale;
    if (mask[ki] <= 0.f) d = -1e9f;
    sc[hd][qi][ki] = d;
  }
  __syncthreads();
  if (t < HEADS * 5) {
    int hd = t / 5, qi = t % 5;
    float mx = -3.4e38f;
    for (int ki = 0; ki < 5; ++ki) mx = fmaxf(mx, sc[hd][qi][ki]);
    float sm = 0.f;
    for (int ki = 0; ki < 5; ++ki) { float e = __expf(sc[hd][qi][ki] - mx); sc[hd][qi][ki] = e; sm += e; }
    float inv = 1.f / sm;
    for (int ki = 0; ki < 5; ++ki) sc[hd][qi][ki] *= inv;
  }
  __syncthreads();
  int hd = t / DH;
  for (int qi = 0; qi < 5; ++qi) {
    float o = 0.f;
    for (int ki = 0; ki < 5; ++ki) o += sc[hd][qi][ki] * v[ki][t];
    O[((long)b * 5 + qi) * HDIM + t] = o;
  }
}

template<int RELU>
__global__ void add_ln_kernel(const float* __restrict__ X, const float* __restrict__ R,
                              const float* __restrict__ g, const float* __restrict__ b,
                              float* __restrict__ out) {
  __shared__ float red[128];
  int r = blockIdx.x, c = threadIdx.x;
  float v = X[(long)r * HDIM + c] + R[(long)r * HDIM + c];
  float m = block_reduce(v, red, c, 128) / 128.f;
  float d = v - m;
  float var = block_reduce(d * d, red, c, 128) / 128.f;
  float y = d * rsqrtf(var + 1e-5f) * g[c] + b[c];
  if (RELU) y = fmaxf(y, 0.f);
  out[(long)r * HDIM + c] = y;
}

__global__ void head2_kernel(const float* __restrict__ Hd, const float* __restrict__ w,
                             const float* __restrict__ b, float* __restrict__ y) {
  __shared__ float red[256];
  int r = blockIdx.x, t = threadIdx.x;
  float tot = block_reduce(Hd[(long)r * 256 + t] * w[t], red, t, 256);
  if (t == 0) y[r] = tot + b[0];
}

__global__ void lstab_final_kernel(const float* __restrict__ scal, float* __restrict__ out) {
  if (threadIdx.x == 0) {
    float l = 0.f;
#pragma unroll
    for (int i = 0; i < 5; ++i) l += fabsf(sqrtf(scal[i]) - sqrtf(scal[5 + i]));
    out[0] = l;
  }
}

// ---------------------------------------------------------------- host orchestration
extern "C" void kernel_launch(void* const* d_in, const int* in_sizes, int n_in,
                              void* d_out, int out_size, void* d_ws, size_t ws_size,
                              hipStream_t stream) {
  (void)in_sizes; (void)n_in; (void)out_size; (void)ws_size;
  const int H = HDIM;
  const float* dem  = (const float*)d_in[0];
  const float* vit  = (const float*)d_in[1];
  const float* itv  = (const float*)d_in[2];
  const float* nstf = (const float*)d_in[3];
  const float* ntsf = (const float*)d_in[4];
  const int*   missing = (const int*)d_in[5];
  const float* pw1 = (const float*)d_in[6];
  const float* pb1 = (const float*)d_in[7];
  const float* pw2 = (const float*)d_in[8];
  const float* pb2 = (const float*)d_in[9];
  const float* pw3 = (const float*)d_in[10];
  const float* pb3 = (const float*)d_in[11];
  const float* bng = (const float*)d_in[12];
  const float* bnb = (const float*)d_in[13];
  const float* simi = (const float*)d_in[14];
  const float* dissim = (const float*)d_in[15];
  const float* gcnw = (const float*)d_in[16];
  const float* gcnb = (const float*)d_in[17];
  const float* adw  = (const float*)d_in[18];
  const float* adb  = (const float*)d_in[19];
  const float* modal = (const float*)d_in[20];
  const float* wq = (const float*)d_in[21];
  const float* bq = (const float*)d_in[22];
  const float* wk = (const float*)d_in[23];
  const float* bk = (const float*)d_in[24];
  const float* wv = (const float*)d_in[25];
  const float* bv = (const float*)d_in[26];
  const float* wo = (const float*)d_in[27];
  const float* bo = (const float*)d_in[28];
  const float* w1t = (const float*)d_in[29];
  const float* b1t = (const float*)d_in[30];
  const float* w2t = (const float*)d_in[31];
  const float* b2t = (const float*)d_in[32];
  const float* lng = (const float*)d_in[33];
  const float* lnb = (const float*)d_in[34];
  const float* ow1 = (const float*)d_in[35];
  const float* ob1 = (const float*)d_in[36];
  const float* ow2 = (const float*)d_in[37];
  const float* ob2 = (const float*)d_in[38];
  float* out = (float*)d_out;

  // --- workspace carve (bump allocator) ---
  char* wp = (char*)d_ws;
  auto alloc = [&](size_t bytes) { char* p = wp; wp += (bytes + 255) & ~(size_t)255; return (void*)p; };
  const long BB = (long)BATCH * BATCH;
  float*  aggf  = (float*)alloc(BB * 4);
  __bf16* aggbf = (__bf16*)alloc(BB * 2);
  float* t0 = (float*)alloc(BH * 4);
  float* t1 = (float*)alloc(BH * 4);
  float* t2 = (float*)alloc(BH * 4);
  float* nste = (float*)alloc(BH * 4);
  float* ntse = (float*)alloc(BH * 4);
  __bf16* y1bf = (__bf16*)alloc(BH * 2);
  __bf16* y2bf = (__bf16*)alloc(BH * 2);
  float* sq1 = (float*)alloc(BATCH * 4);
  float* sq2 = (float*)alloc(BATCH * 4);
  float* meanb = (float*)alloc(H * 4);
  float* rstdb = (float*)alloc(H * 4);
  float* colsum = (float*)alloc(H * 4);
  float* bw1 = (float*)alloc(4 * 4);
  float* bw2 = (float*)alloc(4 * 4);
  float* nstm = (float*)alloc(BATCH * 4);
  float* ntsm = (float*)alloc(BATCH * 4);
  float* scal = (float*)alloc(16 * 4);
  float* xwf  = (float*)alloc(BH * 4);
  __bf16* xwbf = (__bf16*)alloc(BH * 2);
  float* auxA   = (float*)alloc(BH * 4);
  float* nstaux = (float*)alloc(BH * 4);
  float* ntsaux = (float*)alloc(BH * 4);
  float* nstimp = (float*)alloc(BH * 4);
  float* ntsimp = (float*)alloc(BH * 4);
  float* z0 = (float*)alloc(TOKH * 4);
  float* z1 = (float*)alloc(TOKH * 4);
  float* qb = (float*)alloc(TOKH * 4);
  float* kb = (float*)alloc(TOKH * 4);
  float* vb = (float*)alloc(TOKH * 4);
  float* at = (float*)alloc(TOKH * 4);
  float* zl = (float*)alloc(TOKH * 4);
  float* ffh = (float*)alloc((long)TOK * 512 * 4);
  float* headh = (float*)alloc((long)BATCH * 256 * 4);

  // --- masks & masked embeddings ---
  mask_kernel<<<(BATCH + 255) / 256, 256, 0, stream>>>(missing, nstm, ntsm);
  mask_emb_kernel<<<(int)((BH + 255) / 256), 256, 0, stream>>>(nstf, nstm, nste);
  mask_emb_kernel<<<(int)((BH + 255) / 256), 256, 0, stream>>>(ntsf, ntsm, ntse);

  zero_kernel<<<(int)((BB + 255) / 256), 256, 0, stream>>>(aggf, BB);
  zero_kernel<<<1, 256, 0, stream>>>(scal, 16);

  auto run_mlp = [&](const float* base, int p, float* o2) {
    launch_gemm<float, float, 1>(base, pw1 + (long)p * H * H, pb1 + p * H, t0, BATCH, H, H, stream);
    launch_gemm<float, float, 1>(t0,   pw2 + (long)p * H * H, pb2 + p * H, t1, BATCH, H, H, stream);
    launch_gemm<float, float, 0>(t1,   pw3 + (long)p * H * H, pb3 + p * H, o2, BATCH, H, H, stream);
  };

  // --- 5 similarity matrices accumulated into aggf ---
  struct Cfg { const float* base; int proj; int eps; const float* mask; int slot; };
  Cfg cfgs[5] = {
    { dem,  0, 0, nullptr, 0 },
    { vit,  1, 1, nullptr, 1 },
    { itv,  1, 2, nullptr, -1 },  // faithful: itv sim uses proj[1]
    { nste, 3, 3, nstm,    3 },
    { nste, 4, 4, ntsm,    -1 }, // faithful: "nts" sim built from nst_emb
  };
  for (int s = 0; s < 5; ++s) {
    run_mlp(cfgs[s].base, cfgs[s].proj, t2);
    if (cfgs[s].slot >= 0)
      sumsq_kernel<<<512, 256, 0, stream>>>(t2, BH, scal + cfgs[s].slot);
    // wgk features: bn(relu(mlp))
    zero_kernel<<<1, 256, 0, stream>>>(colsum, H);
    bn_stats_kernel<1><<<H, 256, 0, stream>>>(t2, meanb, rstdb);
    bn_norm_kernel<1><<<BATCH, 128, 0, stream>>>(t2, meanb, rstdb, bng, bnb, y1bf, sq1, colsum);
    bw_finalize_kernel<<<1, 256, 0, stream>>>(sq1, colsum, simi, cfgs[s].eps, bw1);
    // gk features: bn(base)
    zero_kernel<<<1, 256, 0, stream>>>(colsum, H);
    bn_stats_kernel<0><<<H, 256, 0, stream>>>(cfgs[s].base, meanb, rstdb);
    bn_norm_kernel<0><<<BATCH, 128, 0, stream>>>(cfgs[s].base, meanb, rstdb, bng, bnb, y2bf, sq2, colsum);
    bw_finalize_kernel<<<1, 256, 0, stream>>>(sq2, colsum, simi, cfgs[s].eps, bw2);
    sim_accum_kernel<<<dim3(BATCH / 128, BATCH / 128), 256, 0, stream>>>(
        y1bf, y2bf, sq1, sq2, bw1, bw2, cfgs[s].mask, aggf);
  }

  // --- remaining lstab terms ---
  run_mlp(itv, 2, t2);
  sumsq_kernel<<<512, 256, 0, stream>>>(t2, BH, scal + 2);
  run_mlp(ntse, 4, t2);
  sumsq_kernel<<<512, 256, 0, stream>>>(t2, BH, scal + 4);
  sumsq_kernel<<<512, 256, 0, stream>>>(dem,  BH, scal + 5);
  sumsq_kernel<<<512, 256, 0, stream>>>(vit,  BH, scal + 6);
  sumsq_kernel<<<512, 256, 0, stream>>>(itv,  BH, scal + 7);
  sumsq_kernel<<<512, 256, 0, stream>>>(nste, BH, scal + 8);
  sumsq_kernel<<<512, 256, 0, stream>>>(ntse, BH, scal + 9);
  lstab_final_kernel<<<1, 32, 0, stream>>>(scal, out + BATCH);

  // --- agg normalize + threshold; bf16 copy for adjacency WMMA GEMMs ---
  finalize_agg_kernel<<<(int)((BB + 255) / 256), 256, 0, stream>>>(aggf, aggbf, nstm, ntsm, dissim);

  // --- 2-layer GCNs (adjacency GEMM: K=4096, A in bf16) ---
  auto gcn_pair = [&](const float* x0, int m, float* outbuf) {
    launch_gemm<float, float, 0>(x0, gcnw + (long)(m * 2 + 0) * H * H, nullptr, xwf, BATCH, H, H, stream);
    cvt_bf16_kernel<<<(int)((BH + 255) / 256), 256, 0, stream>>>(xwf, xwbf, BH);
    launch_gemm<__bf16, __bf16, 1>(aggbf, xwbf, gcnb + (m * 2 + 0) * H, auxA, BATCH, H, BATCH, stream);
    launch_gemm<float, float, 0>(auxA, gcnw + (long)(m * 2 + 1) * H * H, nullptr, xwf, BATCH, H, H, stream);
    cvt_bf16_kernel<<<(int)((BH + 255) / 256), 256, 0, stream>>>(xwf, xwbf, BH);
    launch_gemm<__bf16, __bf16, 1>(aggbf, xwbf, gcnb + (m * 2 + 1) * H, outbuf, BATCH, H, BATCH, stream);
  };
  gcn_pair(nste, 0, nstaux);
  gcn_pair(ntse, 1, ntsaux);

  impute_kernel<<<BATCH, 128, 0, stream>>>(nste, nstaux, nstm, adw + 0 * H, adb + 0, nstimp);
  impute_kernel<<<BATCH, 128, 0, stream>>>(ntse, ntsaux, ntsm, adw + 1 * H, adb + 1, ntsimp);

  // --- transformer over [B,5,H] tokens ---
  build_z0_kernel<<<(int)((TOKH + 255) / 256), 256, 0, stream>>>(dem, vit, itv, nstimp, ntsimp, modal, z0);
  float* zin = z0;
  float* zouts[2] = { z1, z0 };  // layer-1 output reuses z0
  for (int l = 0; l < 2; ++l) {
    launch_gemm<float, float, 0>(zin, wq + (long)l * H * H, bq + l * H, qb, TOK, H, H, stream);
    launch_gemm<float, float, 0>(zin, wk + (long)l * H * H, bk + l * H, kb, TOK, H, H, stream);
    launch_gemm<float, float, 0>(zin, wv + (long)l * H * H, bv + l * H, vb, TOK, H, H, stream);
    if (l == 0) attn_kernel<4><<<BATCH, 128, 0, stream>>>(qb, kb, vb, nstm, ntsm, at);
    else        attn_kernel<1><<<BATCH, 128, 0, stream>>>(qb, kb, vb, nstm, ntsm, at);
    launch_gemm<float, float, 0>(at, wo + (long)l * H * H, bo + l * H, qb, TOK, H, H, stream);
    add_ln_kernel<0><<<TOK, 128, 0, stream>>>(qb, zin, lng + (l * 2 + 0) * H, lnb + (l * 2 + 0) * H, zl);
    launch_gemm<float, float, 1>(zl, w1t + (long)l * H * 4 * H, b1t + l * 4 * H, ffh, TOK, 4 * H, H, stream);
    launch_gemm<float, float, 0>(ffh, w2t + (long)l * 4 * H * H, b2t + l * H, at, TOK, H, 4 * H, stream);
    add_ln_kernel<1><<<TOK, 128, 0, stream>>>(at, zl, lng + (l * 2 + 1) * H, lnb + (l * 2 + 1) * H, zouts[l]);
    zin = zouts[l];
  }

  // --- output head: comb [B,640] -> [B,256] -> [B] ---
  launch_gemm<float, float, 1>(zin, ow1, ob1, headh, BATCH, 2 * H, 5 * H, stream);
  head2_kernel<<<BATCH, 256, 0, stream>>>(headh, ow2, ob2, out);
}
